// MultiHeadAttention_43482248905222
// MI455X (gfx1250) — matrix operand order repaired and checked by reading, not compiled, not run on hardware
//
#include <hip/hip_runtime.h>

// MHA for B=4, T=2048, D=1024, H=16, DK=DV=64 on gfx1250 (wave32, WMMA).
// Pipeline: 3x projection GEMM (f32 in -> bf16 out) -> flash attention
// (bf16 WMMA, f32 online softmax) -> output GEMM (bf16 in -> f32 out).
// Workspace usage: 4 * B*T*D bf16 buffers = 64 MiB of d_ws.
// This revision adds LDS double-buffering (one barrier per K-step) and
// batches WMMA operand loads so the v_wmma ops can issue back-to-back.

typedef __bf16 bf16;
typedef __attribute__((ext_vector_type(16))) __bf16 v16bf;
typedef __attribute__((ext_vector_type(8)))  __bf16 bf16x8;
typedef __attribute__((ext_vector_type(4)))  __bf16 bf16x4;
typedef __attribute__((ext_vector_type(4)))  float  f32x4;
typedef __attribute__((ext_vector_type(8)))  float  v8f;

#define MHA_B 4
#define MHA_T 2048
#define MHA_D 1024
#define MHA_H 16

__device__ __forceinline__ v16bf ld2x8(const bf16* p0, const bf16* p1) {
  union { v16bf v; bf16x8 h[2]; } u;
  u.h[0] = *(const bf16x8*)p0;
  u.h[1] = *(const bf16x8*)p1;
  return u.v;
}
__device__ __forceinline__ v16bf ld16(const bf16* p) { return ld2x8(p, p + 8); }

__device__ __forceinline__ v8f wmma_bf16(v16bf a, v16bf b, v8f c) {
  // D = A(16x32 bf16) * B(32x16 bf16) + C(16x16 f32)
  return __builtin_amdgcn_wmma_f32_16x16x32_bf16(
      false, a, false, b, (short)0, c, false, false);
}

__device__ __forceinline__ float fast_exp2(float x) {
#if __has_builtin(__builtin_amdgcn_exp2f)
  return __builtin_amdgcn_exp2f(x);
#else
  return exp2f(x);
#endif
}

// ---------------------------------------------------------------------------
// Tiled GEMM: Out[M,N] = A[M,K] * W + bias.
// W element (d, j) lives at W[(j>>6)*whs + d*wld + (j&63)], which covers both
// the stacked per-head weights [H,D,64] (whs=D*64, wld=64) and the plain
// row-major Wo [D,D] (whs=64, wld=D).
// Block: 256 threads = 8 waves; tile 128(M) x 64(N); K staged in 32-chunks
// through double-buffered LDS (global loads for chunk k+1 are in flight
// while chunk k is consumed by WMMA).
// ---------------------------------------------------------------------------
template <bool A_BF16, bool OUT_F32>
__global__ __launch_bounds__(256)
void gemm_bias_kernel(const void* __restrict__ Aptr,
                      const float* __restrict__ W,
                      const float* __restrict__ bias,
                      void* __restrict__ Out,
                      int M, int N, int Kd, int whs, int wld) {
  __shared__ __align__(16) bf16 lA[2][128 * 32];   // [m][kk], 2 x 8 KB
  __shared__ __align__(16) bf16 lB[2][64 * 32];    // [n][kk] transposed, 2 x 4 KB

  const int tid  = threadIdx.x;
  const int lane = tid & 31;
  const int wv   = tid >> 5;                       // wave 0..7 -> 16-row slab
  const int m0   = blockIdx.x * 128;
  const int n0   = blockIdx.y * 64;

  const float* Af = (const float*)Aptr;
  const bf16*  Ab = (const bf16*)Aptr;

  v8f acc0 = {}, acc1 = {}, acc2 = {}, acc3 = {};

  const int lrow = lane & 15;
  const int kb = (lane >> 4) * 8;    // unused; A is also loaded contiguous at ks
  const int ks = (lane >> 4) * 16;   // B-operand contiguous 16-start
  const int arow = wv * 16 + lrow;

  // Register staging for the next K-chunk (double buffering).
  bf16x4 ra[4];
  f32x4  rb[2];

  auto loadChunk = [&](int k0) {
#pragma unroll
    for (int i = 0; i < 4; ++i) {                  // A tile 128x32
      int e4 = tid + i * 256;
      int m = e4 >> 3, kk = (e4 & 7) * 4;
      size_t g = (size_t)(m0 + m) * Kd + (k0 + kk);
      if (A_BF16) {
        ra[i] = *(const bf16x4*)&Ab[g];
      } else {
        f32x4 x = *(const f32x4*)&Af[g];
        ra[i][0] = (bf16)x[0]; ra[i][1] = (bf16)x[1];
        ra[i][2] = (bf16)x[2]; ra[i][3] = (bf16)x[3];
      }
    }
#pragma unroll
    for (int i = 0; i < 2; ++i) {                  // W tile 32x64 (n-fast)
      int e4 = tid + i * 256;
      int kk = e4 >> 4, n = (e4 & 15) * 4;
      int j = n0 + n;
      rb[i] = *(const f32x4*)&W[(size_t)(j >> 6) * whs +
                                (size_t)(k0 + kk) * wld + (j & 63)];
    }
  };

  auto storeChunk = [&](int buf) {
#pragma unroll
    for (int i = 0; i < 4; ++i) {
      int e4 = tid + i * 256;
      int m = e4 >> 3, kk = (e4 & 7) * 4;
      *(bf16x4*)&lA[buf][m * 32 + kk] = ra[i];
    }
#pragma unroll
    for (int i = 0; i < 2; ++i) {
      int e4 = tid + i * 256;
      int kk = e4 >> 4, n = (e4 & 15) * 4;
#pragma unroll
      for (int c = 0; c < 4; ++c) lB[buf][(n + c) * 32 + kk] = (bf16)rb[i][c];
    }
  };

  loadChunk(0);
  storeChunk(0);
  __syncthreads();

  const int nsteps = Kd >> 5;
  for (int it = 0; it < nsteps; ++it) {
    const int cur = it & 1, nxt = cur ^ 1;
    const bool hasNext = (it + 1) < nsteps;
    if (hasNext) loadChunk((it + 1) * 32);   // global loads in flight

    // Batch all operand loads, then issue 4 WMMAs back-to-back.
    const bf16* ar = &lA[cur][arow * 32];
    v16bf a  = ld16(ar + ks);
    v16bf b0 = ld16(&lB[cur][( 0 + lrow) * 32 + ks]);
    v16bf b1 = ld16(&lB[cur][(16 + lrow) * 32 + ks]);
    v16bf b2 = ld16(&lB[cur][(32 + lrow) * 32 + ks]);
    v16bf b3 = ld16(&lB[cur][(48 + lrow) * 32 + ks]);
    acc0 = wmma_bf16(a, b0, acc0);
    acc1 = wmma_bf16(a, b1, acc1);
    acc2 = wmma_bf16(a, b2, acc2);
    acc3 = wmma_bf16(a, b3, acc3);

    if (hasNext) storeChunk(nxt);            // fill the other buffer
    __syncthreads();                         // one barrier per K-step
  }

  // Epilogue: C layout -> lane holds column (lane&15), rows r(+8 for hi half).
  const int rbase = m0 + wv * 16 + (lane >> 4) * 8;
#pragma unroll
  for (int nt = 0; nt < 4; ++nt) {
    v8f acc = (nt == 0) ? acc0 : (nt == 1) ? acc1 : (nt == 2) ? acc2 : acc3;
    int col = n0 + nt * 16 + lrow;
    float bval = bias[col];
#pragma unroll
    for (int r = 0; r < 8; ++r) {
      float vv = acc[r] + bval;
      size_t o = (size_t)(rbase + r) * N + col;
      if (OUT_F32) ((float*)Out)[o] = vv;
      else         ((bf16*)Out)[o]  = (bf16)vv;
    }
  }
}

// ---------------------------------------------------------------------------
// Flash attention over one (b, h, 64 q-rows) per 128-thread block.
// Each wave owns a 16-row Q tile; K/V blocks of 32 keys are staged through
// double-buffered LDS (V transposed so the PV B-operand is contiguous).
// The per-wave P reshape buffer needs no barrier: LDS ops from one wave
// complete in order, so same-wave cross-lane exchange is safe.
// Online softmax in exp2 domain with the 1/sqrt(dk) scale folded in.
// ---------------------------------------------------------------------------
__global__ __launch_bounds__(128)
void attn_kernel(const bf16* __restrict__ Q, const bf16* __restrict__ K,
                 const bf16* __restrict__ V, bf16* __restrict__ O) {
  __shared__ __align__(16) bf16 sK[2][32 * 64];    // [s][d]   2 x 4 KB
  __shared__ __align__(16) bf16 sV[2][64 * 32];    // [v][s]   2 x 4 KB
  __shared__ __align__(16) bf16 sP[4][16 * 32];    // per-wave P tiles, 4 KB

  const int tid  = threadIdx.x;
  const int lane = tid & 31;
  const int wv   = tid >> 5;            // 0..3
  const int qb   = blockIdx.x;          // T/64 blocks
  const int h    = blockIdx.y;
  const int b    = blockIdx.z;

  const size_t base = ((size_t)b * MHA_T) * MHA_D + (size_t)h * 64;
  const int t0   = qb * 64 + wv * 16;
  const int lrow = lane & 15;
  const int kb   = (lane >> 4) * 8;
  const int ks   = (lane >> 4) * 16;

  // Q A-operands for d 0:32 and 32:64 (rows are contiguous bf16 in memory).
  const bf16* qrow = Q + base + (size_t)(t0 + lrow) * MHA_D;
  const v16bf aQ0 = ld16(qrow + ks);
  const v16bf aQ1 = ld16(qrow + 32 + ks);

  const float sc = 0.125f * 1.44269504f;  // 1/sqrt(64) * log2(e)

  float mi[8], li[8];
#pragma unroll
  for (int r = 0; r < 8; ++r) { mi[r] = -1e30f; li[r] = 0.0f; }
  v8f o0 = {}, o1 = {}, o2 = {}, o3 = {};

  // Cooperative K/V staging (V transposed), double buffered.
  const int srow = tid >> 2;            // 0..31
  const int d0   = (tid & 3) * 16;
  bf16x8 kr0, kr1, vr0, vr1;

  auto loadKV = [&](int s0) {
    const bf16* kp = K + base + (size_t)(s0 + srow) * MHA_D + d0;
    kr0 = *(const bf16x8*)kp;
    kr1 = *(const bf16x8*)(kp + 8);
    const bf16* vp = V + base + (size_t)(s0 + srow) * MHA_D + d0;
    vr0 = *(const bf16x8*)vp;
    vr1 = *(const bf16x8*)(vp + 8);
  };
  auto storeKV = [&](int buf) {
    *(bf16x8*)&sK[buf][srow * 64 + d0]     = kr0;
    *(bf16x8*)&sK[buf][srow * 64 + d0 + 8] = kr1;
#pragma unroll
    for (int j = 0; j < 8; ++j) {
      sV[buf][(d0 + j) * 32 + srow]     = vr0[j];
      sV[buf][(d0 + 8 + j) * 32 + srow] = vr1[j];
    }
  };

  loadKV(0);
  storeKV(0);
  __syncthreads();

  const int nsteps = MHA_T / 32;
  for (int it = 0; it < nsteps; ++it) {
    const int cur = it & 1, nxt = cur ^ 1;
    const bool hasNext = (it + 1) < nsteps;
    if (hasNext) loadKV((it + 1) * 32);   // next block's global loads in flight

    // S = Q * K^T, two 16-key subtiles; batch operands, then 4 WMMAs.
    const bf16* k0r = &sK[cur][( 0 + lrow) * 64];
    const bf16* k1r = &sK[cur][(16 + lrow) * 64];
    v16bf bK00 = ld16(k0r + ks), bK01 = ld16(k0r + 32 + ks);
    v16bf bK10 = ld16(k1r + ks), bK11 = ld16(k1r + 32 + ks);
    v8f s0a = {}, s1a = {};
    s0a = wmma_bf16(aQ0, bK00, s0a);
    s1a = wmma_bf16(aQ0, bK10, s1a);
    s0a = wmma_bf16(aQ1, bK01, s0a);
    s1a = wmma_bf16(aQ1, bK11, s1a);

    v8f s_t[2];
#pragma unroll
    for (int r = 0; r < 8; ++r) { s_t[0][r] = s0a[r] * sc; s_t[1][r] = s1a[r] * sc; }

    // Online softmax. C layout: lane = column, lanes 0-15 rows 0-7,
    // lanes 16-31 rows 8-15 -> reduce across 16-lane halves.
#pragma unroll
    for (int r = 0; r < 8; ++r) {
      float rm = fmaxf(s_t[0][r], s_t[1][r]);
#pragma unroll
      for (int off = 8; off > 0; off >>= 1)
        rm = fmaxf(rm, __shfl_xor(rm, off, 16));
      float mn = fmaxf(mi[r], rm);
      float alpha = fast_exp2(mi[r] - mn);
      float p0 = fast_exp2(s_t[0][r] - mn);
      float p1 = fast_exp2(s_t[1][r] - mn);
      float rs = p0 + p1;
#pragma unroll
      for (int off = 8; off > 0; off >>= 1)
        rs += __shfl_xor(rs, off, 16);
      li[r] = li[r] * alpha + rs;
      mi[r] = mn;
      o0[r] *= alpha; o1[r] *= alpha; o2[r] *= alpha; o3[r] *= alpha;
      int prow = r + (lane >> 4) * 8;
      sP[wv][prow * 32 + lrow]      = (bf16)p0;
      sP[wv][prow * 32 + 16 + lrow] = (bf16)p1;
    }
    // Per-wave C-layout -> A-layout reshape through LDS (in-order per wave,
    // no workgroup barrier needed).
    const bf16* pr = &sP[wv][lrow * 32];
    v16bf aP  = ld16(pr + ks);
    v16bf bV0 = ld16(&sV[cur][( 0 + lrow) * 32 + ks]);
    v16bf bV1 = ld16(&sV[cur][(16 + lrow) * 32 + ks]);
    v16bf bV2 = ld16(&sV[cur][(32 + lrow) * 32 + ks]);
    v16bf bV3 = ld16(&sV[cur][(48 + lrow) * 32 + ks]);
    o0 = wmma_bf16(aP, bV0, o0);
    o1 = wmma_bf16(aP, bV1, o1);
    o2 = wmma_bf16(aP, bV2, o2);
    o3 = wmma_bf16(aP, bV3, o3);

    if (hasNext) storeKV(nxt);            // fill the other buffer
    __syncthreads();                      // one barrier per key-block
  }

  // Normalize and write head output straight into concat layout [B*T, H*64].
#pragma unroll
  for (int r = 0; r < 8; ++r) {
    float inv = 1.0f / li[r];
    int trow = t0 + r + (lane >> 4) * 8;
    bf16* orow = O + base + (size_t)trow * MHA_D;
    orow[ 0 + lrow] = (bf16)(o0[r] * inv);
    orow[16 + lrow] = (bf16)(o1[r] * inv);
    orow[32 + lrow] = (bf16)(o2[r] * inv);
    orow[48 + lrow] = (bf16)(o3[r] * inv);
  }
}

extern "C" void kernel_launch(void* const* d_in, const int* in_sizes, int n_in,
                              void* d_out, int out_size, void* d_ws, size_t ws_size,
                              hipStream_t stream) {
  const float* q  = (const float*)d_in[0];
  const float* k  = (const float*)d_in[1];
  const float* v  = (const float*)d_in[2];
  const float* Wq = (const float*)d_in[3];
  const float* bq = (const float*)d_in[4];
  const float* Wk = (const float*)d_in[5];
  const float* bk = (const float*)d_in[6];
  const float* Wv = (const float*)d_in[7];
  const float* bv = (const float*)d_in[8];
  const float* Wo = (const float*)d_in[9];
  const float* bo = (const float*)d_in[10];

  const int M  = MHA_B * MHA_T;   // 8192
  const int N  = MHA_D;           // 1024
  const int Kd = MHA_D;           // 1024

  // Workspace: Q,K,V projections + concat buffer, all bf16 [M, D]. 64 MiB.
  bf16* Qb = (bf16*)d_ws;
  bf16* Kb = Qb + (size_t)M * N;
  bf16* Vb = Kb + (size_t)M * N;
  bf16* Cc = Vb + (size_t)M * N;

  dim3 gg(M / 128, N / 64), gb(256);
  // Per-head stacked weights: whs = D*64, wld = 64.
  gemm_bias_kernel<false, false><<<gg, gb, 0, stream>>>(q, Wq, bq, Qb, M, N, Kd, MHA_D * 64, 64);
  gemm_bias_kernel<false, false><<<gg, gb, 0, stream>>>(k, Wk, bk, Kb, M, N, Kd, MHA_D * 64, 64);
  gemm_bias_kernel<false, false><<<gg, gb, 0, stream>>>(v, Wv, bv, Vb, M, N, Kd, MHA_D * 64, 64);

  attn_kernel<<<dim3(MHA_T / 64, MHA_H, MHA_B), dim3(128), 0, stream>>>(Qb, Kb, Vb, Cc);

  // Output projection: plain row-major Wo -> whs = 64, wld = D. f32 out.
  gemm_bias_kernel<true, true><<<gg, gb, 0, stream>>>(Cc, Wo, bo, (float*)d_out, M, N, Kd, 64, MHA_D);
}
